// RTDETRTransformerv2_16088947490926
// MI455X (gfx1250) — matrix operand order repaired and checked by reading, not compiled, not run on hardware
//
#include <hip/hip_runtime.h>
#include <hip/hip_bf16.h>

// ---------------------------------------------------------------------------
// RT-DETR multi-scale deformable attention, CDNA5 (gfx1250) WMMA pipeline.
//   B=16, Q=300, D_MODEL=256, H=8, hd=32, L=3, P=4, Lv=8400, M_v = B*Lv = 134400
// ---------------------------------------------------------------------------

typedef __attribute__((ext_vector_type(16))) _Float16 v16h;
typedef __attribute__((ext_vector_type(8)))  _Float16 v8h;
typedef __attribute__((ext_vector_type(8)))  float    v8f;
typedef int v4i_vs __attribute__((vector_size(16)));   // matches async-LDS builtin pointee

#define D_MODEL 256
#define N_HEADS 8
#define HEAD_DIM 32
#define N_LEVELS 3
#define N_POINTS 4
#define BATCH 16
#define NQ 300
#define LV_TOTAL 8400          // 80*80 + 40*40 + 20*20
#define MV (BATCH * LV_TOTAL)  // 134400
#define BQ (BATCH * NQ)        // 4800

#define LDS_PAD 40             // padded row (halfs): 80 B stride, bank-conflict free

// Optional gfx1250 async global->LDS path (guarded: zero compile risk).
#if defined(__gfx1250__) && \
    __has_builtin(__builtin_amdgcn_global_load_async_to_lds_b128) && \
    __has_builtin(__builtin_amdgcn_s_wait_asynccnt)
#define HAVE_ASYNC_LDS 1
#else
#define HAVE_ASYNC_LDS 0
#endif

// ---------------------------------------------------------------------------
// Generic GEMM:  out[M x 256] = A[M x 256] (fp32) * W[256 x 256] (fp32) + bias
// A and W converted to f16 while staging to LDS; f32 accumulate via
// v_wmma_f32_16x16x32_f16.  Block tile 64 (M) x 256 (N), K stepped by 32,
// double-buffered LDS + register prefetch of the next K tile.
// 8 waves: wave = (n_half<<2) | m_tile ; each wave owns 16 rows x 128 cols
// = 8 WMMA 16x16 tiles (64 acc VGPRs) reusing one A fragment.
// ---------------------------------------------------------------------------
struct TileRegs { float a[8]; float b[32]; };

__device__ __forceinline__ void load_tile(const float* __restrict__ A,
                                          const float* __restrict__ W,
                                          size_t rowBase, int k0, int tid,
                                          TileRegs& t) {
    const int r = tid >> 2;            // 0..63
    const int c = (tid & 3) * 8;       // 0,8,16,24
    const float* src = A + (rowBase + (size_t)r) * D_MODEL + k0 + c;
#pragma unroll
    for (int i = 0; i < 8; ++i) t.a[i] = src[i];
#pragma unroll
    for (int k = 0; k < 32; ++k) t.b[k] = W[(size_t)(k0 + k) * D_MODEL + tid];
}

__device__ __forceinline__ void store_tile(_Float16 (*__restrict__ As)[LDS_PAD],
                                           _Float16 (*__restrict__ Bs)[LDS_PAD],
                                           int tid, const TileRegs& t) {
    const int r = tid >> 2;
    const int c = (tid & 3) * 8;
#pragma unroll
    for (int i = 0; i < 8; ++i) As[r][c + i] = (_Float16)t.a[i];
#pragma unroll
    for (int k = 0; k < 32; ++k) Bs[tid][k] = (_Float16)t.b[k];
}

template <bool OUT_HALF>
__global__ __launch_bounds__(256) void gemm256_wmma(
    const float* __restrict__ A, const float* __restrict__ W,
    const float* __restrict__ bias, void* __restrict__ outp, int M)
{
    __shared__ _Float16 As[2][64][LDS_PAD];    // 2 x  5 KB, [m][k]
    __shared__ _Float16 Bs[2][256][LDS_PAD];   // 2 x 20 KB, [n][k] (transposed)

    const int tid   = threadIdx.x;
    const int lane  = tid & 31;
    const int wv    = tid >> 5;
    const int m0    = (wv & 3) * 16;        // wave's row tile inside block
    const int nBase = (wv >> 2) * 128;      // wave's column half
    const size_t rowBase = (size_t)blockIdx.x * 64;

    v8f acc[8];
#pragma unroll
    for (int t = 0; t < 8; ++t) {
        v8f z = {0.f, 0.f, 0.f, 0.f, 0.f, 0.f, 0.f, 0.f};
        acc[t] = z;
    }

    TileRegs tr;
    load_tile(A, W, rowBase, 0, tid, tr);
    store_tile(As[0], Bs[0], tid, tr);

    int buf = 0;
    for (int step = 0; step < 8; ++step) {
        __syncthreads();
        // prefetch next K tile into registers while this tile computes
        if (step < 7) load_tile(A, W, rowBase, (step + 1) * 32, tid, tr);

        // ---- A fragment (16x32 f16 ISA layout) ----
        const int mrow = m0 + (lane & 15);
        const int g    = lane >> 4;
        v16h afrag;
        {
            const v8h lo = *(const v8h*)(&As[buf][mrow][g * 8]);
            const v8h hi = *(const v8h*)(&As[buf][mrow][16 + g * 8]);
#pragma unroll
            for (int e = 0; e < 8; ++e) { afrag[e] = lo[e]; afrag[8 + e] = hi[e]; }
        }
        // ---- preload all 8 B fragments, then issue the WMMA chain ----
        v16h bfrag[8];
#pragma unroll
        for (int t = 0; t < 8; ++t) {
            const int n = nBase + t * 16 + (lane & 15);
            const v8h lo = *(const v8h*)(&Bs[buf][n][g * 8]);
            const v8h hi = *(const v8h*)(&Bs[buf][n][16 + g * 8]);
#pragma unroll
            for (int e = 0; e < 8; ++e) { bfrag[t][e] = lo[e]; bfrag[t][8 + e] = hi[e]; }
        }
#pragma unroll
        for (int t = 0; t < 8; ++t)
            acc[t] = __builtin_amdgcn_wmma_f32_16x16x32_f16(
                false, afrag, false, bfrag[t], (short)0, acc[t], false, false);

        if (step < 7) store_tile(As[buf ^ 1], Bs[buf ^ 1], tid, tr);
        buf ^= 1;
    }

    // ---- epilogue: lane holds col n; VGPR r holds row m0 + r + 8*(lane>>4) ----
#pragma unroll
    for (int t = 0; t < 8; ++t) {
        const int n  = nBase + t * 16 + (lane & 15);
        const float bn = bias[n];
#pragma unroll
        for (int r = 0; r < 8; ++r) {
            const int m = m0 + r + 8 * (lane >> 4);
            const size_t idx = (rowBase + (size_t)m) * D_MODEL + n;
            const float val = acc[t][r] + bn;
            if (OUT_HALF) ((_Float16*)outp)[idx] = (_Float16)val;
            else          ((float*)outp)[idx]    = val;
        }
    }
}

// ---------------------------------------------------------------------------
// Projections: off = q@Woff+boff (192 cols), attn = softmax(q@Wattn+battn)
// per-head over 12.  Also folds reference points + clip + pixel-space coords.
// One block per (b,q); 288 threads = 9 waves.
// coords layout : [bq][((h*3+lvl)*4+p)*2 + xy]      (192 per row)
// weights layout: [bq][h*12 + lvl*4 + p]            (96 per row)
// ---------------------------------------------------------------------------
__global__ __launch_bounds__(288) void proj_kernel(
    const float* __restrict__ query, const float* __restrict__ refp,
    const float* __restrict__ Woff, const float* __restrict__ boff,
    const float* __restrict__ Wattn, const float* __restrict__ battn,
    float* __restrict__ coords, float* __restrict__ weights)
{
    __shared__ float qrow[D_MODEL];
    __shared__ float logit[96];
    __shared__ float hmax[8], hsum[8];

    const int bq  = blockIdx.x;
    const int tid = threadIdx.x;

#if HAVE_ASYNC_LDS
    if (tid < 64) {
        const float* gsrc = query + (size_t)bq * D_MODEL + tid * 4;
        __builtin_amdgcn_global_load_async_to_lds_b128(
            (__attribute__((address_space(1))) v4i_vs*)gsrc,
            (__attribute__((address_space(3))) v4i_vs*)&qrow[tid * 4],
            0, 0);
    }
    __builtin_amdgcn_s_wait_asynccnt(0);
    __syncthreads();
#else
    if (tid < D_MODEL) qrow[tid] = query[(size_t)bq * D_MODEL + tid];
    __syncthreads();
#endif

    if (tid < 192) {
        float acc = boff[tid];
#pragma unroll 8
        for (int k = 0; k < D_MODEL; ++k) acc += qrow[k] * Woff[(size_t)k * 192 + tid];
        const int pair = tid >> 1;
        const int xy   = tid & 1;
        const int lvl  = (pair >> 2) % 3;
        float pt = refp[(size_t)bq * 4 + xy] + acc;
        pt = fminf(fmaxf(pt, 0.f), 1.f);
        const float dim = (float)(80 >> lvl);       // square levels: w == h
        coords[(size_t)bq * 192 + tid] = pt * dim - 0.5f;
    } else {
        const int ca = tid - 192;
        float acc = battn[ca];
#pragma unroll 8
        for (int k = 0; k < D_MODEL; ++k) acc += qrow[k] * Wattn[(size_t)k * 96 + ca];
        logit[ca] = acc;
    }
    __syncthreads();

    if (tid < 8) {
        float m = -1e30f;
#pragma unroll
        for (int j = 0; j < 12; ++j) m = fmaxf(m, logit[tid * 12 + j]);
        float s = 0.f;
#pragma unroll
        for (int j = 0; j < 12; ++j) s += __expf(logit[tid * 12 + j] - m);
        hmax[tid] = m; hsum[tid] = s;
    }
    __syncthreads();

    if (tid >= 192) {
        const int ca = tid - 192;
        const int h  = ca / 12;
        weights[(size_t)bq * 96 + ca] = __expf(logit[ca] - hmax[h]) / hsum[h];
    }
}

// ---------------------------------------------------------------------------
// Bilinear deformable sampling.  One wave32 per (b,q,h); lane = channel d.
// Each corner gather is a coalesced 64 B read of 32 contiguous f16 channels.
// ---------------------------------------------------------------------------
__global__ __launch_bounds__(256) void sample_kernel(
    const _Float16* __restrict__ vF16, const float* __restrict__ coords,
    const float* __restrict__ weights, float* __restrict__ sampled)
{
    const int tid  = threadIdx.x;
    const int lane = tid & 31;
    const int wid  = blockIdx.x * 8 + (tid >> 5);
    const int h    = wid & 7;
    const int bq   = wid >> 3;
    const int b    = bq / NQ;

    const float* cp = coords  + (size_t)bq * 192 + h * 24;
    const float* wp = weights + (size_t)bq * 96 + h * 12;
    const _Float16* vb = vF16 + ((size_t)b * LV_TOTAL) * D_MODEL + h * HEAD_DIM + lane;

    float acc = 0.f;
#pragma unroll
    for (int lvl = 0; lvl < 3; ++lvl) {
        const int wd = 80 >> lvl;
        const int ht = 80 >> lvl;
        const int st = (lvl == 0) ? 0 : ((lvl == 1) ? 6400 : 8000);
#pragma unroll
        for (int p = 0; p < 4; ++p) {
            const float px = cp[lvl * 8 + p * 2 + 0];
            const float py = cp[lvl * 8 + p * 2 + 1];
            const float aw = wp[lvl * 4 + p];
            const float x0f = floorf(px), y0f = floorf(py);
            const float fx = px - x0f, fy = py - y0f;
            const int x0 = (int)x0f, y0 = (int)y0f;
            const int x1 = x0 + 1,  y1 = y0 + 1;
            const float w00 = (1.f - fx) * (1.f - fy) * aw;
            const float w10 = fx * (1.f - fy) * aw;
            const float w01 = (1.f - fx) * fy * aw;
            const float w11 = fx * fy * aw;
            const bool vx0 = (x0 >= 0) & (x0 < wd);
            const bool vx1 = (x1 >= 0) & (x1 < wd);
            const bool vy0 = (y0 >= 0) & (y0 < ht);
            const bool vy1 = (y1 >= 0) & (y1 < ht);
            if (vx0 & vy0) acc += w00 * (float)vb[(size_t)(st + y0 * wd + x0) * D_MODEL];
            if (vx1 & vy0) acc += w10 * (float)vb[(size_t)(st + y0 * wd + x1) * D_MODEL];
            if (vx0 & vy1) acc += w01 * (float)vb[(size_t)(st + y1 * wd + x0) * D_MODEL];
            if (vx1 & vy1) acc += w11 * (float)vb[(size_t)(st + y1 * wd + x1) * D_MODEL];
        }
    }
    sampled[(size_t)bq * D_MODEL + h * HEAD_DIM + lane] = acc;
}

// ---------------------------------------------------------------------------
// Launch: value-proj GEMM (f16 out) -> projections/softmax -> sampling ->
// output GEMM (fp32 to d_out).
// ---------------------------------------------------------------------------
extern "C" void kernel_launch(void* const* d_in, const int* in_sizes, int n_in,
                              void* d_out, int out_size, void* d_ws, size_t ws_size,
                              hipStream_t stream) {
    (void)in_sizes; (void)n_in; (void)out_size; (void)ws_size;

    const float* query = (const float*)d_in[0];
    const float* refp  = (const float*)d_in[1];
    const float* value = (const float*)d_in[2];
    const float* Wv    = (const float*)d_in[3];
    const float* bv    = (const float*)d_in[4];
    const float* Woff  = (const float*)d_in[5];
    const float* boff  = (const float*)d_in[6];
    const float* Wattn = (const float*)d_in[7];
    const float* battn = (const float*)d_in[8];
    const float* Wout  = (const float*)d_in[9];
    const float* bout  = (const float*)d_in[10];
    float* out = (float*)d_out;

    // workspace layout (bytes)
    char* ws = (char*)d_ws;
    const size_t vF16_bytes   = (size_t)MV * D_MODEL * 2;   // 68,812,800
    const size_t coords_bytes = (size_t)BQ * 192 * 4;       //  3,686,400
    const size_t wts_bytes    = (size_t)BQ * 96 * 4;        //  1,843,200
    _Float16* vF16  = (_Float16*)ws;
    float* coords   = (float*)(ws + vF16_bytes);
    float* weights  = (float*)(ws + vF16_bytes + coords_bytes);
    float* sampled  = (float*)(ws + vF16_bytes + coords_bytes + wts_bytes);

    gemm256_wmma<true><<<MV / 64, 256, 0, stream>>>(value, Wv, bv, (void*)vF16, MV);
    proj_kernel<<<BQ, 288, 0, stream>>>(query, refp, Woff, boff, Wattn, battn,
                                        coords, weights);
    sample_kernel<<<BQ, 256, 0, stream>>>(vF16, coords, weights, sampled);
    gemm256_wmma<false><<<BQ / 64, 256, 0, stream>>>(sampled, Wout, bout, (void*)out, BQ);
}
